// LSS_48584670052371
// MI455X (gfx1250) — compile-verified, matrix operand order checked by reading, not run on hardware
//
#include <hip/hip_runtime.h>

typedef float v8f __attribute__((ext_vector_type(8)));
typedef float v2f __attribute__((ext_vector_type(2)));
typedef _Float16 v16h __attribute__((ext_vector_type(16)));

#define B_   2
#define N_   6
#define D_   60
#define FH_  28
#define FW_  60
#define C_   64
#define NX_  200
#define NY_  200
#define HW_  (FH_ * FW_)          /* 1680 */
#define NXY_ (NX_ * NY_)          /* 40000 */
#define PIX_ (B_ * N_ * FH_ * FW_) /* 20160 pixels */
#define OUT_ (B_ * C_ * NXY_)     /* 5,120,000 floats */
#define NMAT_ (B_ * N_)           /* 12 camera matrices */

// ---------------------------------------------------------------------------
// Kernel 0: zero the output (harness poisons d_out with 0xAA before timing).
// ---------------------------------------------------------------------------
__global__ __launch_bounds__(256) void lss_zero_out(float4* __restrict__ out, int n4) {
    int i = blockIdx.x * blockDim.x + threadIdx.x;
    if (i < n4) out[i] = make_float4(0.f, 0.f, 0.f, 0.f);
}

// ---------------------------------------------------------------------------
// Kernel 1: invert the 12 lidar2img 4x4 matrices -> img2lidar in workspace.
// Adjugate/cofactor method (layout-agnostic: works row-major in == row-major out).
// ---------------------------------------------------------------------------
__global__ void lss_invert_mats(const float* __restrict__ l2i, float* __restrict__ i2l) {
    int mi = threadIdx.x;
    if (mi >= NMAT_) return;
    const float* m = l2i + mi * 16;
    float inv[16];
    inv[0]  =  m[5]*m[10]*m[15] - m[5]*m[11]*m[14] - m[9]*m[6]*m[15] + m[9]*m[7]*m[14] + m[13]*m[6]*m[11] - m[13]*m[7]*m[10];
    inv[4]  = -m[4]*m[10]*m[15] + m[4]*m[11]*m[14] + m[8]*m[6]*m[15] - m[8]*m[7]*m[14] - m[12]*m[6]*m[11] + m[12]*m[7]*m[10];
    inv[8]  =  m[4]*m[9]*m[15]  - m[4]*m[11]*m[13] - m[8]*m[5]*m[15] + m[8]*m[7]*m[13] + m[12]*m[5]*m[11] - m[12]*m[7]*m[9];
    inv[12] = -m[4]*m[9]*m[14]  + m[4]*m[10]*m[13] + m[8]*m[5]*m[14] - m[8]*m[6]*m[13] - m[12]*m[5]*m[10] + m[12]*m[6]*m[9];
    inv[1]  = -m[1]*m[10]*m[15] + m[1]*m[11]*m[14] + m[9]*m[2]*m[15] - m[9]*m[3]*m[14] - m[13]*m[2]*m[11] + m[13]*m[3]*m[10];
    inv[5]  =  m[0]*m[10]*m[15] - m[0]*m[11]*m[14] - m[8]*m[2]*m[15] + m[8]*m[3]*m[14] + m[12]*m[2]*m[11] - m[12]*m[3]*m[10];
    inv[9]  = -m[0]*m[9]*m[15]  + m[0]*m[11]*m[13] + m[8]*m[1]*m[15] - m[8]*m[3]*m[13] - m[12]*m[1]*m[11] + m[12]*m[3]*m[9];
    inv[13] =  m[0]*m[9]*m[14]  - m[0]*m[10]*m[13] - m[8]*m[1]*m[14] + m[8]*m[2]*m[13] + m[12]*m[1]*m[10] - m[12]*m[2]*m[9];
    inv[2]  =  m[1]*m[6]*m[15]  - m[1]*m[7]*m[14]  - m[5]*m[2]*m[15] + m[5]*m[3]*m[14] + m[13]*m[2]*m[7]  - m[13]*m[3]*m[6];
    inv[6]  = -m[0]*m[6]*m[15]  + m[0]*m[7]*m[14]  + m[4]*m[2]*m[15] - m[4]*m[3]*m[14] - m[12]*m[2]*m[7]  + m[12]*m[3]*m[6];
    inv[10] =  m[0]*m[5]*m[15]  - m[0]*m[7]*m[13]  - m[4]*m[1]*m[15] + m[4]*m[3]*m[13] + m[12]*m[1]*m[7]  - m[12]*m[3]*m[5];
    inv[14] = -m[0]*m[5]*m[14]  + m[0]*m[6]*m[13]  + m[4]*m[1]*m[14] - m[4]*m[2]*m[13] - m[12]*m[1]*m[6]  + m[12]*m[2]*m[5];
    inv[3]  = -m[1]*m[6]*m[11]  + m[1]*m[7]*m[10]  + m[5]*m[2]*m[11] - m[5]*m[3]*m[10] - m[9]*m[2]*m[7]   + m[9]*m[3]*m[6];
    inv[7]  =  m[0]*m[6]*m[11]  - m[0]*m[7]*m[10]  - m[4]*m[2]*m[11] + m[4]*m[3]*m[10] + m[8]*m[2]*m[7]   - m[8]*m[3]*m[6];
    inv[11] = -m[0]*m[5]*m[11]  + m[0]*m[7]*m[9]   + m[4]*m[1]*m[11] - m[4]*m[3]*m[9]  - m[8]*m[1]*m[7]   + m[8]*m[3]*m[5];
    inv[15] =  m[0]*m[5]*m[10]  - m[0]*m[6]*m[9]   - m[4]*m[1]*m[10] + m[4]*m[2]*m[9]  + m[8]*m[1]*m[6]   - m[8]*m[2]*m[5];
    float det = m[0]*inv[0] + m[1]*inv[4] + m[2]*inv[8] + m[3]*inv[12];
    float r = 1.0f / det;
    float* o = i2l + mi * 16;
    #pragma unroll
    for (int k = 0; k < 16; ++k) o[k] = inv[k] * r;
}

// ---------------------------------------------------------------------------
// Kernel 2: fused softmax + WMMA rank-1 lift + geometry + voxel scatter.
// One wave32 per pixel; 4 waves (128 threads) per block.
// ---------------------------------------------------------------------------
__global__ __launch_bounds__(128) void lss_splat(
    const float* __restrict__ feats,    // (BN, C, FH, FW)
    const float* __restrict__ depth,    // (BN, D, FH, FW)
    const float* __restrict__ frustum,  // (D, FH, FW, 3)
    const float* __restrict__ i2l,      // (BN, 4, 4) inverted
    const float* __restrict__ dxv,      // (3,)
    const float* __restrict__ bxv,      // (3,)
    float* __restrict__ out)            // (B, C, NX, NY)
{
    __shared__ float wts[4][64];  // softmax weights per wave, padded D 60->64
    __shared__ float fts[4][64];  // feats per wave

    const int lane = threadIdx.x & 31;
    const int wv   = threadIdx.x >> 5;
    const int pi   = blockIdx.x * 4 + wv;   // pixel index; PIX_ % 4 == 0

    const int w  = pi % FW_;
    const int h  = (pi / FW_) % FH_;
    const int bn = pi / (FW_ * FH_);
    const int b  = bn / N_;

    // ---- softmax over D=60 depth bins (wave32 butterfly reductions) ----
    const float* dp = depth + (long)bn * D_ * HW_ + h * FW_ + w;
    const int d1 = lane + 32;
    float v0 = dp[lane * HW_];                                   // d = 0..31
    float v1 = (d1 < D_) ? dp[d1 * HW_] : -__builtin_inff();     // d = 32..59
    float mx = fmaxf(v0, v1);
    #pragma unroll
    for (int off = 16; off > 0; off >>= 1) mx = fmaxf(mx, __shfl_xor(mx, off, 32));
    float e0 = __expf(v0 - mx);
    float e1 = (d1 < D_) ? __expf(v1 - mx) : 0.f;
    float s = e0 + e1;
    #pragma unroll
    for (int off = 16; off > 0; off >>= 1) s += __shfl_xor(s, off, 32);
    const float rs = 1.f / s;
    wts[wv][lane]      = e0 * rs;
    wts[wv][lane + 32] = e1 * rs;   // pad bins 60..63 get exactly 0

    // ---- stage the 64 channel features for this pixel ----
    const float* fp = feats + (long)bn * C_ * HW_ + h * FW_ + w;
    fts[wv][lane]      = fp[lane * HW_];
    fts[wv][lane + 32] = fp[(lane + 32) * HW_];
    __syncthreads();

    // ---- geometry: img2lidar rows for this camera ----
    const float* M = i2l + (long)bn * 16;
    const float m00 = M[0], m01 = M[1], m02 = M[2],  m03 = M[3];
    const float m10 = M[4], m11 = M[5], m12 = M[6],  m13 = M[7];
    const float m20 = M[8], m21 = M[9], m22 = M[10], m23 = M[11];
    const float dx0 = dxv[0], dx1 = dxv[1], dx2 = dxv[2];
    const float lb0 = bxv[0] - 0.5f * dx0;
    const float lb1 = bxv[1] - 0.5f * dx1;
    const float lb2 = bxv[2] - 0.5f * dx2;

    // per-lane voxel offset for d = 16*j + (lane&15); -1 if dropped
    const int dl = lane & 15;
    int voff[4];
    #pragma unroll
    for (int j = 0; j < 4; ++j) {
        const int d = j * 16 + dl;
        const bool in_d = (d < D_);
        float u = 0.f, vv = 0.f, dz = 1.f;
        if (in_d) {
            const float* fr = frustum + ((long)d * FH_ + h) * FW_ * 3 + (long)w * 3;
            u = fr[0]; vv = fr[1]; dz = fr[2];
        }
        const float dd = fmaxf(dz, 1e-5f);
        const float p0 = u * dd, p1 = vv * dd, p2 = dz;
        const float gx = m00 * p0 + m01 * p1 + m02 * p2 + m03;
        const float gy = m10 * p0 + m11 * p1 + m12 * p2 + m13;
        const float gz = m20 * p0 + m21 * p1 + m22 * p2 + m23;
        // trunc-toward-zero matches astype(int32)
        const int ix = (int)((gx - lb0) / dx0);
        const int iy = (int)((gy - lb1) / dx1);
        const int iz = (int)((gz - lb2) / dx2);
        const bool kept = in_d && (ix >= 0) && (ix < NX_) && (iy >= 0) && (iy < NY_) && (iz == 0);
        voff[j] = kept ? (ix * NY_ + iy) : -1;
    }

    float* outb = out + (long)b * C_ * NXY_;
    const int half = lane >> 4;           // 0: rows 0..7, 1: rows 8..15 of tile

    // ---- rank-1 lift via WMMA: x[c,d] = feat[c] * w[d], 4x4 tiles of 16x16 ----
    #pragma unroll
    for (int i = 0; i < 4; ++i) {
        float av = fts[wv][i * 16 + dl];
        av = (lane < 16) ? av : 0.f;      // only K=0 slot populated
        #pragma unroll
        for (int j = 0; j < 4; ++j) {
            float bv = wts[wv][j * 16 + dl];
            bv = (lane < 16) ? bv : 0.f;
            v8f acc = {};
#if __has_builtin(__builtin_amdgcn_wmma_f32_16x16x4_f32)
            // A 16x4 f32 (col 0 = feats chunk), B 4x16 f32 (row 0 = weights chunk)
            v2f af = {av, 0.f};
            v2f bf = {bv, 0.f};
            acc = __builtin_amdgcn_wmma_f32_16x16x4_f32(
                false, af, false, bf, (short)0, acc, false, false);
#else
            // Fallback: f16 16x16x32 with only halfword K=0 populated
            v16h ah = {}; ah[0] = (_Float16)av;
            v16h bh = {}; bh[0] = (_Float16)bv;
            acc = __builtin_amdgcn_wmma_f32_16x16x32_f16(
                false, ah, false, bh, (short)0, acc, false, false);
#endif
            // D layout: lane&15 = column n (depth bin), VGPR r = row m = r + 8*half
            const int vo = voff[j];
            if (vo >= 0) {
                float* p = outb + vo + (long)(i * 16 + half * 8) * NXY_;
                #pragma unroll
                for (int r = 0; r < 8; ++r)
                    atomicAdd(p + (long)r * NXY_, acc[r]);
            }
        }
    }
}

// ---------------------------------------------------------------------------
extern "C" void kernel_launch(void* const* d_in, const int* in_sizes, int n_in,
                              void* d_out, int out_size, void* d_ws, size_t ws_size,
                              hipStream_t stream) {
    const float* feats   = (const float*)d_in[0];
    const float* depth   = (const float*)d_in[1];
    const float* l2i     = (const float*)d_in[2];
    const float* frustum = (const float*)d_in[3];
    const float* dx      = (const float*)d_in[4];
    const float* bx      = (const float*)d_in[5];
    float* out = (float*)d_out;
    float* i2l = (float*)d_ws;   // 12 * 16 floats

    lss_zero_out<<<(OUT_ / 4 + 255) / 256, 256, 0, stream>>>((float4*)out, OUT_ / 4);
    lss_invert_mats<<<1, 32, 0, stream>>>(l2i, i2l);
    lss_splat<<<PIX_ / 4, 128, 0, stream>>>(feats, depth, frustum, i2l, dx, bx, out);
}
